// SquashedGaussianQuantMLPActor_15994458210910
// MI455X (gfx1250) — matrix operand order repaired and checked by reading, not compiled, not run on hardware
//
#include <hip/hip_runtime.h>

// ---------------------------------------------------------------------------
// Types for CDNA5 WMMA (wave32): V_WMMA_F32_16X16X32_BF16
// ---------------------------------------------------------------------------
typedef __bf16 bf16;
typedef __attribute__((ext_vector_type(16))) __bf16 v16bf;
typedef __attribute__((ext_vector_type(8)))  float  v8f;
typedef __attribute__((ext_vector_type(4)))  int    v4i;

#define S_IN (1.0f/12000.0f)
#define AS_GLOBAL __attribute__((address_space(1)))
#define AS_LDS    __attribute__((address_space(3)))

// ---------------------------------------------------------------------------
// CDNA5 async global->LDS staging (GLOBAL_LOAD_ASYNC_TO_LDS_B128, ASYNCcnt)
// with graceful fallback to sync vector copies if the builtin is absent.
// Probe result: param0 = int4 addrspace(1)*, param1 = int4 addrspace(3)*.
// ---------------------------------------------------------------------------
#if defined(__gfx1250__) && __has_builtin(__builtin_amdgcn_global_load_async_to_lds_b128)
#define HAVE_ASYNC_LDS 1
#else
#define HAVE_ASYNC_LDS 0
#endif

__device__ __forceinline__ void stage16(const bf16* g, bf16* s) {
#if HAVE_ASYNC_LDS
  __builtin_amdgcn_global_load_async_to_lds_b128(
      (AS_GLOBAL v4i*)g, (AS_LDS v4i*)s, 0, 0);
#else
  *(uint4*)s = *(const uint4*)g;
#endif
}

__device__ __forceinline__ void stage_wait() {
#if HAVE_ASYNC_LDS
#if __has_builtin(__builtin_amdgcn_s_wait_asynccnt)
  __builtin_amdgcn_s_wait_asynccnt(0);
#else
  asm volatile("s_wait_asynccnt 0" ::: "memory");
#endif
#endif
}

// ---------------------------------------------------------------------------
// Small prep kernels: absmax reduction, weight/bias fake-quant, obs->bf16
// ---------------------------------------------------------------------------
__global__ void init_slots_kernel(unsigned* slots) {
  if (threadIdx.x < 3) slots[threadIdx.x] = 0u;
}

__global__ void absmax_kernel(const float* __restrict__ w, int n, unsigned* slot) {
  __shared__ float red[256];
  float m = 0.0f;
  for (int i = blockIdx.x * blockDim.x + threadIdx.x; i < n; i += gridDim.x * blockDim.x)
    m = fmaxf(m, fabsf(w[i]));
  red[threadIdx.x] = m;
  __syncthreads();
  for (int s = 128; s > 0; s >>= 1) {
    if ((int)threadIdx.x < s) red[threadIdx.x] = fmaxf(red[threadIdx.x], red[threadIdx.x + s]);
    __syncthreads();
  }
  if (threadIdx.x == 0) atomicMax(slot, __float_as_uint(red[0]));
}

__global__ void quant_w_kernel(const float* __restrict__ w, bf16* __restrict__ out,
                               int n, const unsigned* __restrict__ slot) {
  const float amax  = __uint_as_float(*slot);
  const float s     = amax * (1.0f / 127.0f);
  const float inv_s = 127.0f / amax;
  for (int i = blockIdx.x * blockDim.x + threadIdx.x; i < n; i += gridDim.x * blockDim.x) {
    float q = rintf(w[i] * inv_s);                 // round-half-even (RNE mode)
    q = fminf(fmaxf(q, -127.0f), 127.0f);
    out[i] = (bf16)(q * s);
  }
}

__global__ void quant_b_kernel(const float* __restrict__ b1, const float* __restrict__ b2,
                               const float* __restrict__ b3,
                               float* __restrict__ b1q, float* __restrict__ b2q,
                               float* __restrict__ b3q,
                               const unsigned* __restrict__ slots) {
  const float ws1 = __uint_as_float(slots[0]) * (1.0f / 127.0f);
  const float ws2 = __uint_as_float(slots[1]) * (1.0f / 127.0f);
  const float ws3 = __uint_as_float(slots[2]) * (1.0f / 127.0f);
  const float sc1 = S_IN * ws1;   // bias scale fc1 = in_scale * w_scale
  const float sc2 = sc1  * ws2;   // act scale passes through ReLU
  const float sc3 = sc2  * ws3;
  for (int i = threadIdx.x; i < 1024; i += blockDim.x)
    b1q[i] = fminf(fmaxf(rintf(b1[i] / sc1), -128.0f), 127.0f) * sc1;
  for (int i = threadIdx.x; i < 1024; i += blockDim.x)
    b2q[i] = fminf(fmaxf(rintf(b2[i] / sc2), -128.0f), 127.0f) * sc2;
  for (int i = threadIdx.x; i < 128; i += blockDim.x)
    b3q[i] = fminf(fmaxf(rintf(b3[i] / sc3), -128.0f), 127.0f) * sc3;
}

__global__ void obs_to_bf16_kernel(const float* __restrict__ obs, bf16* __restrict__ x, int n) {
  const float c0 = S_IN * S_IN;  // quantize_input_tensor scaling
  for (int i = blockIdx.x * blockDim.x + threadIdx.x; i < n; i += gridDim.x * blockDim.x)
    x[i] = (bf16)(obs[i] * c0);
}

// ---------------------------------------------------------------------------
// bf16 WMMA GEMM:  out[M,N] = A[M,K] @ W[N,K]^T + bias, optional ReLU
// Block tile 128x128, BK=32. 8 waves/block, each wave owns 32(M)x64(N):
// 2x4 tiles of v_wmma_f32_16x16x32_bf16 (f32 accum).
// LDS tiles row-major, pitch 48 elems (96B rows: 16B-aligned, bank-staggered).
// Staging: fully unrolled, pointer-increment addressing, async->LDS if present.
// Fragment addressing follows the CDNA5 ISA 16-bit A(16x32)/B(32x16) layouts.
// ---------------------------------------------------------------------------
union FragBF { uint4 q[2]; v16bf v; };

template <bool RELU, bool F32OUT>
__global__ __launch_bounds__(256) void gemm_bf16_wmma(
    const bf16* __restrict__ A, const bf16* __restrict__ W,
    const float* __restrict__ bias,
    bf16* __restrict__ outB, float* __restrict__ outF,
    int M, int N, int K) {
  constexpr int PITCH = 48;              // 32 + 16 pad elems; 96B rows, 16B aligned
  __shared__ bf16 lA[128 * PITCH];
  __shared__ bf16 lB[128 * PITCH];

  const int tid  = threadIdx.x;
  const int lane = tid & 31;
  const int wave = tid >> 5;
  const int wr   = wave & 3;             // 4 wave-rows  * 32 M
  const int wc   = wave >> 2;            // 2 wave-cols  * 64 N
  const int half = lane >> 4;            // lane group (ISA: K-halves)
  const int r    = lane & 15;

  const int rowBase = blockIdx.y * 128;
  const int colBase = blockIdx.x * 128;

  // --- staging addresses: each thread moves 4x16B per K step --------------
  const int srow = tid >> 2;             // tile row 0..63 (and +64)
  const int sc8  = (tid & 3) << 3;       // K-chunk offset 0/8/16/24
  const bf16* gA0 = A + (size_t)(rowBase + srow) * K + sc8;
  const bf16* gA1 = gA0 + (size_t)64 * K;
  const bf16* gB0 = W + (size_t)(colBase + srow) * K + sc8;
  const bf16* gB1 = gB0 + (size_t)64 * K;
  bf16* sA0 = &lA[srow * PITCH + sc8];
  bf16* sA1 = sA0 + 64 * PITCH;
  bf16* sB0 = &lB[srow * PITCH + sc8];
  bf16* sB1 = sB0 + 64 * PITCH;

  // --- fragment LDS read addresses (loop-invariant) ------------------------
  const bf16* fA0 = &lA[(wr * 32 + r) * PITCH + half * 8];
  const bf16* fB0 = &lB[(wc * 64 + r) * PITCH + half * 8];

  v8f acc[2][4] = {};

  for (int k0 = 0; k0 < K; k0 += 32) {
    stage16(gA0, sA0);
    stage16(gA1, sA1);
    stage16(gB0, sB0);
    stage16(gB1, sB1);
    if (k0 + 32 < K) {                   // global_prefetch_b8 of next K tile
      __builtin_prefetch(gA0 + 32, 0, 3);
      __builtin_prefetch(gA1 + 32, 0, 3);
      __builtin_prefetch(gB0 + 32, 0, 3);
      __builtin_prefetch(gB1 + 32, 0, 3);
    }
    gA0 += 32; gA1 += 32; gB0 += 32; gB1 += 32;
    stage_wait();                        // s_wait_asynccnt 0 (own transfers)
    __syncthreads();                     // all waves' LDS writes visible

    // A fragment (16x32): lanes 0-15 hold K{0..7,16..23}, lanes 16-31 K{8..15,24..31}.
    FragBF af[2], bfr[4];
#pragma unroll
    for (int mi = 0; mi < 2; ++mi) {
      const bf16* p = fA0 + mi * 16 * PITCH;
      af[mi].q[0] = *(const uint4*)(p);
      af[mi].q[1] = *(const uint4*)(p + 16);
    }
    // B fragment (32x16): lanes 0-15 hold K{0..7,16..23}, lanes 16-31 K{8..15,24..31} (col = r).
#pragma unroll
    for (int ni = 0; ni < 4; ++ni) {
      const bf16* p = fB0 + ni * 16 * PITCH;
      bfr[ni].q[0] = *(const uint4*)(p);
      bfr[ni].q[1] = *(const uint4*)(p + 16);
    }
#pragma unroll
    for (int mi = 0; mi < 2; ++mi)
#pragma unroll
      for (int ni = 0; ni < 4; ++ni)
        acc[mi][ni] = __builtin_amdgcn_wmma_f32_16x16x32_bf16(
            false, af[mi].v, false, bfr[ni].v, (short)0, acc[mi][ni], false, false);
    __syncthreads();
  }

  // C/D layout: VGPR v in lane l -> M = v + 8*(l>>4), N = l&15.
#pragma unroll
  for (int mi = 0; mi < 2; ++mi) {
#pragma unroll
    for (int ni = 0; ni < 4; ++ni) {
      const int gcol = colBase + wc * 64 + ni * 16 + r;
      const float bv = bias[gcol];
#pragma unroll
      for (int v = 0; v < 8; ++v) {
        const int grow = rowBase + wr * 32 + mi * 16 + half * 8 + v;
        float val = acc[mi][ni][v] + bv;
        if (RELU) val = fmaxf(val, 0.0f);
        if (F32OUT) outF[(size_t)grow * N + gcol] = val;
        else        outB[(size_t)grow * N + gcol] = (bf16)val;
      }
    }
  }
}

// ---------------------------------------------------------------------------
// Actor head: mu/log_std split, rsample, tanh-squash, logp reduce (64 dims).
// One row per block, 64 threads (2 waves).
// ---------------------------------------------------------------------------
__global__ void actor_head_kernel(const float* __restrict__ net,
                                  const float* __restrict__ eps,
                                  float* __restrict__ action,
                                  float* __restrict__ logp) {
  const int row = blockIdx.x;
  const int j   = threadIdx.x;  // 0..63
  const float mu = net[(size_t)row * 128 + j];
  float ls = net[(size_t)row * 128 + 64 + j];
  ls = fminf(fmaxf(ls, -20.0f), 2.0f);
  const float sd = __expf(ls);
  const float pi = fmaf(sd, eps[(size_t)row * 64 + j], mu);
  const float t  = (pi - mu) / sd;
  float lp = -0.5f * t * t - ls - 0.91893853320467274178f;     // 0.5*log(2*pi)
  const float z  = -2.0f * pi;
  const float sp = fmaxf(z, 0.0f) + log1pf(__expf(-fabsf(z))); // softplus(-2*pi)
  lp -= 2.0f * (0.69314718055994530942f - pi - sp);

  __shared__ float red[64];
  red[j] = lp;
  __syncthreads();
  for (int s = 32; s > 0; s >>= 1) {
    if (j < s) red[j] += red[j + s];
    __syncthreads();
  }
  action[(size_t)row * 64 + j] = (10000.0f * tanhf(pi)) * (1.0f / 10000.0f);
  if (j == 0) logp[row] = red[0];
}

// ---------------------------------------------------------------------------
// Host launcher
// ---------------------------------------------------------------------------
extern "C" void kernel_launch(void* const* d_in, const int* in_sizes, int n_in,
                              void* d_out, int out_size, void* d_ws, size_t ws_size,
                              hipStream_t stream) {
  (void)in_sizes; (void)n_in; (void)out_size; (void)ws_size;
  const int B = 32768, OBS = 512, H = 1024, ACT = 64, NOUT = 2 * ACT;

  const float* obs = (const float*)d_in[0];
  const float* eps = (const float*)d_in[1];
  const float* W1  = (const float*)d_in[2];
  const float* b1  = (const float*)d_in[3];
  const float* W2  = (const float*)d_in[4];
  const float* b2  = (const float*)d_in[5];
  const float* W3  = (const float*)d_in[6];
  const float* b3  = (const float*)d_in[7];

  float* action = (float*)d_out;                       // [B, ACT]
  float* logp   = (float*)d_out + (size_t)B * ACT;     // [B]

  char* ws = (char*)d_ws;
  size_t off = 0;
  auto alloc = [&](size_t bytes) -> void* {
    void* p = ws + off;
    off = (off + bytes + 255) & ~(size_t)255;
    return p;
  };
  unsigned* slots = (unsigned*)alloc(3 * sizeof(unsigned));
  float* b1q = (float*)alloc((size_t)H    * sizeof(float));
  float* b2q = (float*)alloc((size_t)H    * sizeof(float));
  float* b3q = (float*)alloc((size_t)NOUT * sizeof(float));
  bf16*  w1q = (bf16*)alloc((size_t)H    * OBS * sizeof(bf16));
  bf16*  w2q = (bf16*)alloc((size_t)H    * H   * sizeof(bf16));
  bf16*  w3q = (bf16*)alloc((size_t)NOUT * H   * sizeof(bf16));
  bf16*  xb  = (bf16*)alloc((size_t)B * OBS * sizeof(bf16));
  bf16*  h1  = (bf16*)alloc((size_t)B * H   * sizeof(bf16));
  bf16*  h2  = (bf16*)alloc((size_t)B * H   * sizeof(bf16));
  float* net = (float*)alloc((size_t)B * NOUT * sizeof(float));

  init_slots_kernel<<<dim3(1), dim3(32), 0, stream>>>(slots);
  absmax_kernel<<<dim3(256), dim3(256), 0, stream>>>(W1, H * OBS,    slots + 0);
  absmax_kernel<<<dim3(256), dim3(256), 0, stream>>>(W2, H * H,      slots + 1);
  absmax_kernel<<<dim3(256), dim3(256), 0, stream>>>(W3, NOUT * H,   slots + 2);
  quant_w_kernel<<<dim3(512), dim3(256), 0, stream>>>(W1, w1q, H * OBS,  slots + 0);
  quant_w_kernel<<<dim3(512), dim3(256), 0, stream>>>(W2, w2q, H * H,    slots + 1);
  quant_w_kernel<<<dim3(512), dim3(256), 0, stream>>>(W3, w3q, NOUT * H, slots + 2);
  quant_b_kernel<<<dim3(1), dim3(256), 0, stream>>>(b1, b2, b3, b1q, b2q, b3q, slots);
  obs_to_bf16_kernel<<<dim3(2048), dim3(256), 0, stream>>>(obs, xb, B * OBS);

  dim3 blk(256);
  gemm_bf16_wmma<true,  false><<<dim3(H / 128,    B / 128), blk, 0, stream>>>(
      xb, w1q, b1q, h1, nullptr, B, H, OBS);
  gemm_bf16_wmma<true,  false><<<dim3(H / 128,    B / 128), blk, 0, stream>>>(
      h1, w2q, b2q, h2, nullptr, B, H, H);
  gemm_bf16_wmma<false, true ><<<dim3(NOUT / 128, B / 128), blk, 0, stream>>>(
      h2, w3q, b3q, nullptr, net, B, NOUT, H);

  actor_head_kernel<<<dim3(B), dim3(64), 0, stream>>>(net, eps, action, logp);
}
